// NMT_40432822125208
// MI455X (gfx1250) — matrix operand order repaired and checked by reading, not compiled, not run on hardware
//
#include <hip/hip_runtime.h>
#include <math.h>

// ---------------- problem constants ----------------
#define SRC_LEN 50
#define TRG_LEN 40
#define BATCH   64
#define ENC_H   512
#define DEC_H   1024
#define EMB     300
#define EMB_P   320            // EMB padded to multiple of 32
#define TRG_V   23262
#define V_P     23264          // TRG_V padded to multiple of 16
#define K_DEC_IN 1344          // (DEC_H + EMB) padded to multiple of 64

// ---------------- WMMA types ----------------
typedef __attribute__((ext_vector_type(16))) __bf16 v16bf;
typedef __attribute__((ext_vector_type(8)))  __bf16 v8bf;
typedef __attribute__((ext_vector_type(8)))  float  v8f;

union BF16Frag { v8bf h[2]; v16bf v; };

__device__ __forceinline__ float sigf(float x) { return 1.0f / (1.0f + __expf(-x)); }

// Async copy of 16 bytes/lane from global to LDS (ASYNCcnt-tracked, gfx1250)
#define ASYNC_LDS_B128(laddr, gptr)                                           \
  asm volatile("global_load_async_to_lds_b128 %0, %1, off"                    \
               :: "v"(laddr), "v"(gptr) : "memory")
#define WAIT_ASYNC(n) asm volatile("s_wait_asynccnt " #n ::: "memory")

#define WMMA_BF16(a, b, c)                                                    \
  __builtin_amdgcn_wmma_f32_16x16x32_bf16(false, (a), false, (b), (short)0,   \
                                          (c), false, false)

// ---------------------------------------------------------------------------
// bf16 GEMM:  C(MxN,f32) = A(MxK,row-major) @ W^T, W stored (N x K) row-major.
// One wave owns a 64x16 C strip: 4 accumulators share one B fragment, so each
// B element is fetched exactly once (min L2 traffic for the streaming weight).
// B strip (16 rows x 64 halves = 2KB/stage) is double-buffered in wave-private
// LDS via GLOBAL_LOAD_ASYNC_TO_LDS_B128. A fragments are software-pipelined
// in a0/a1 ping-pong register sets: the next k-step's fragments are loaded
// BEFORE the current WMMA group and stay live across it, so global loads
// overlap XDL execution instead of serializing per-WMMA.
// Requires M%64==0, N%16==0, K%64==0.
// ---------------------------------------------------------------------------
__global__ __launch_bounds__(128) void k_gemm_bf16(
    const __bf16* __restrict__ A, int lda,
    const __bf16* __restrict__ Bm, int ldb,
    float* __restrict__ C, int ldc,
    int N, int K) {
  const int lane = threadIdx.x & 31;
  const int wave = threadIdx.x >> 5;
  const int ntile = blockIdx.x * 4 + wave;
  if (ntile * 16 >= N) return;            // wave-uniform guard (EXEC stays full)
  const int mblk = blockIdx.y;            // 64 rows of C per block-row

  // wave-private double buffer: 2 x (16 rows x 64 halves) = 4KB
  __shared__ alignas(256) __bf16 smem[4 * 2 * 16 * 64];
  __bf16* wbuf = smem + wave * (2 * 16 * 64);
  const unsigned wbuf_addr = (unsigned)(size_t)wbuf;   // low 32 bits = LDS offset

  // --- staging: per-lane flattened tile offset o = i*512 + lane*16 bytes ---
  const char* Bbase = (const char*)Bm + (size_t)(ntile * 16) * ((size_t)ldb * 2);
  const long  ldbB  = (long)ldb * 2;
#define STAGE(s, buf)                                                          \
  {                                                                            \
    _Pragma("unroll")                                                          \
    for (int i = 0; i < 4; ++i) {                                              \
      int o = i * 512 + lane * 16;         /* byte in 2KB tile image */        \
      const char* g = Bbase + (long)(o >> 7) * ldbB + (long)(s) * 128 + (o & 127); \
      unsigned l = wbuf_addr + (unsigned)((buf) * 2048 + o);                   \
      ASYNC_LDS_B128(l, g);                                                    \
    }                                                                          \
  }

  // --- A fragment pointers: 4 M-tiles, lane layout per 16x32 bf16 A spec ---
  const int akb = (lane >> 4) << 3;        // 0 or 8
  const __bf16* Ap[4];
#pragma unroll
  for (int mt = 0; mt < 4; ++mt)
    Ap[mt] = A + (size_t)(mblk * 64 + mt * 16 + (lane & 15)) * lda + akb;

  // --- B fragment LDS offsets (halves): row = lane&15, kb = (lane>>4)*8 ---
  const int bfrag_half = (lane & 15) * 64 + ((lane >> 4) << 3);

  v8f acc[4];
#pragma unroll
  for (int mt = 0; mt < 4; ++mt)
    acc[mt] = (v8f){0.f, 0.f, 0.f, 0.f, 0.f, 0.f, 0.f, 0.f};

  const int nstages = K >> 6;              // 64 halves of K per stage
  STAGE(0, 0);
  if (nstages > 1) STAGE(1, 1);

  // prologue: a0 holds k=0 fragments
  BF16Frag a0[4], a1[4];
#pragma unroll
  for (int mt = 0; mt < 4; ++mt) {
    a0[mt].h[0] = *(const v8bf*)(Ap[mt]);
    a0[mt].h[1] = *(const v8bf*)(Ap[mt] + 16);
  }

  for (int s = 0; s < nstages; ++s) {
    if (s + 1 < nstages) { WAIT_ASYNC(4); } else { WAIT_ASYNC(0); }
    const __bf16* bb = wbuf + (s & 1) * 1024 + bfrag_half;

    // both 32-deep B fragments of this stage from LDS
    BF16Frag b0, b1;
    b0.h[0] = *(const v8bf*)(bb);
    b0.h[1] = *(const v8bf*)(bb + 16);
    b1.h[0] = *(const v8bf*)(bb + 32);
    b1.h[1] = *(const v8bf*)(bb + 48);

    // prefetch k = s*64+32 fragments; live across the first WMMA group
    const int k1 = s * 64 + 32;
#pragma unroll
    for (int mt = 0; mt < 4; ++mt) {
      a1[mt].h[0] = *(const v8bf*)(Ap[mt] + k1);
      a1[mt].h[1] = *(const v8bf*)(Ap[mt] + k1 + 16);
    }

#pragma unroll
    for (int mt = 0; mt < 4; ++mt)
      acc[mt] = WMMA_BF16(a0[mt].v, b0.v, acc[mt]);

    // prefetch next stage's k = (s+1)*64 fragments during the second group
    if (s + 1 < nstages) {
      const int k2 = (s + 1) * 64;
#pragma unroll
      for (int mt = 0; mt < 4; ++mt) {
        a0[mt].h[0] = *(const v8bf*)(Ap[mt] + k2);
        a0[mt].h[1] = *(const v8bf*)(Ap[mt] + k2 + 16);
      }
    }

#pragma unroll
    for (int mt = 0; mt < 4; ++mt)
      acc[mt] = WMMA_BF16(a1[mt].v, b1.v, acc[mt]);

    if (s + 2 < nstages) STAGE(s + 2, s & 1);   // safe: stage s consumed above
  }

  const int cc = ntile * 16 + (lane & 15);
#pragma unroll
  for (int mt = 0; mt < 4; ++mt) {
    const int r = mblk * 64 + mt * 16 + ((lane >> 4) << 3);
    float* Cp = C + (size_t)r * ldc + cc;
#pragma unroll
    for (int i = 0; i < 8; ++i) Cp[(size_t)i * ldc] = acc[mt][i];
  }
#undef STAGE
}

// ---------------------------------------------------------------------------
// f32 -> bf16 cast with row/col zero padding (weights; also pads gen_W rows)
// ---------------------------------------------------------------------------
__global__ void k_cast_pad(const float* __restrict__ src, __bf16* __restrict__ dst,
                           int src_rows, int src_cols, int dst_rows, int dst_cols) {
  long i = (long)blockIdx.x * blockDim.x + threadIdx.x;
  long total = (long)dst_rows * dst_cols;
  if (i >= total) return;
  int r = (int)(i / dst_cols), c = (int)(i % dst_cols);
  float v = (r < src_rows && c < src_cols) ? src[(long)r * src_cols + c] : 0.f;
  dst[i] = (__bf16)v;
}

__global__ void k_fill_f32(float* p, long n, float v) {
  long i = (long)blockIdx.x * blockDim.x + threadIdx.x;
  if (i < n) p[i] = v;
}
__global__ void k_fill_bf16(__bf16* p, long n, float v) {
  long i = (long)blockIdx.x * blockDim.x + threadIdx.x;
  if (i < n) p[i] = (__bf16)v;
}

// Gather source embeddings -> Xsrc bf16 (SRC_LEN*BATCH, EMB_P)
__global__ void k_embed_src(const int* __restrict__ tok, const float* __restrict__ emb,
                            __bf16* __restrict__ X) {
  long i = (long)blockIdx.x * blockDim.x + threadIdx.x;
  long total = (long)SRC_LEN * BATCH * EMB_P;
  if (i >= total) return;
  int row = (int)(i / EMB_P), e = (int)(i % EMB_P);
  float v = (e < EMB) ? emb[(long)tok[row] * EMB + e] : 0.f;
  X[i] = (__bf16)v;
}

// Encoder LSTM gate fusion (one direction), writes h (bf16), c (f32), enc_out slice
__global__ void k_enc_gate(const float* __restrict__ XW, int xw_row0,
                           const float* __restrict__ HW,
                           const float* __restrict__ bih, const float* __restrict__ bhh,
                           float* __restrict__ c, __bf16* __restrict__ h,
                           float* __restrict__ enc_out, int t, int col_off) {
  int i = blockIdx.x * blockDim.x + threadIdx.x;
  if (i >= BATCH * ENC_H) return;
  int b = i / ENC_H, j = i % ENC_H;
  const float* xg = XW + (long)(xw_row0 + b) * (4 * ENC_H);
  const float* hg = HW + (long)b * (4 * ENC_H);
  float gi = xg[j]             + hg[j]             + bih[j]             + bhh[j];
  float gf = xg[j + ENC_H]     + hg[j + ENC_H]     + bih[j + ENC_H]     + bhh[j + ENC_H];
  float gg = xg[j + 2 * ENC_H] + hg[j + 2 * ENC_H] + bih[j + 2 * ENC_H] + bhh[j + 2 * ENC_H];
  float go = xg[j + 3 * ENC_H] + hg[j + 3 * ENC_H] + bih[j + 3 * ENC_H] + bhh[j + 3 * ENC_H];
  float cn = sigf(gf) * c[i] + sigf(gi) * tanhf(gg);
  float hn = sigf(go) * tanhf(cn);
  c[i] = cn;
  h[i] = (__bf16)hn;
  enc_out[((long)t * BATCH + b) * (2 * ENC_H) + col_off + j] = hn;
}

// Attention: scores/softmax/context for one batch row per block.
// Writes st (bf16) into A_wo[:, 0:DEC_H].
__global__ void k_attention(const float* __restrict__ enc_out,
                            const float* __restrict__ q,
                            __bf16* __restrict__ A_wo) {
  const int b = blockIdx.x;
  const int tid = threadIdx.x, nt = blockDim.x;   // 256
  __shared__ float red[256];
  __shared__ float sc[SRC_LEN];
  __shared__ float alpha[SRC_LEN];
  const float* qb = q + (long)b * DEC_H;
  for (int s = 0; s < SRC_LEN; ++s) {
    const float* eo = enc_out + ((long)s * BATCH + b) * DEC_H;
    float p = 0.f;
    for (int d = tid; d < DEC_H; d += nt) p += eo[d] * qb[d];
    red[tid] = p; __syncthreads();
    for (int o = nt >> 1; o > 0; o >>= 1) {
      if (tid < o) red[tid] += red[tid + o];
      __syncthreads();
    }
    if (tid == 0) sc[s] = red[0];
    __syncthreads();
  }
  if (tid == 0) {
    float m = -1e30f;
    for (int s = 0; s < SRC_LEN; ++s) m = fmaxf(m, sc[s]);
    float sum = 0.f;
    for (int s = 0; s < SRC_LEN; ++s) { alpha[s] = __expf(sc[s] - m); sum += alpha[s]; }
    float inv = 1.f / sum;
    for (int s = 0; s < SRC_LEN; ++s) alpha[s] *= inv;
  }
  __syncthreads();
  for (int d = tid; d < DEC_H; d += nt) {
    float st = 0.f;
    for (int s = 0; s < SRC_LEN; ++s)
      st += alpha[s] * enc_out[((long)s * BATCH + b) * DEC_H + d];
    A_wo[(long)b * (2 * DEC_H) + d] = (__bf16)st;
  }
}

// tanh(ct_lin) and decoder-LSTM input assembly: inp = [ct | emb_prev | pad]
__global__ void k_ct_inp(const float* __restrict__ ct_lin,
                         const float* __restrict__ dec_emb,
                         const int* __restrict__ trg, int step,
                         __bf16* __restrict__ inp) {
  int i = blockIdx.x * blockDim.x + threadIdx.x;
  if (i >= BATCH * K_DEC_IN) return;
  int b = i / K_DEC_IN, d = i % K_DEC_IN;
  float v;
  if (d < DEC_H)               v = tanhf(ct_lin[(long)b * DEC_H + d]);
  else if (d < DEC_H + EMB)    v = dec_emb[(long)trg[step * BATCH + b] * EMB + (d - DEC_H)];
  else                         v = 0.f;
  inp[i] = (__bf16)v;
}

// Decoder LSTM gate fusion; writes h (bf16), c (f32), and h into A_wo[:, DEC_H:]
__global__ void k_dec_lstm(const float* __restrict__ G1, const float* __restrict__ G2,
                           const float* __restrict__ bih, const float* __restrict__ bhh,
                           float* __restrict__ c, __bf16* __restrict__ h,
                           __bf16* __restrict__ A_wo) {
  int i = blockIdx.x * blockDim.x + threadIdx.x;
  if (i >= BATCH * DEC_H) return;
  int b = i / DEC_H, j = i % DEC_H;
  const float* g1 = G1 + (long)b * 4 * DEC_H;
  const float* g2 = G2 + (long)b * 4 * DEC_H;
  float gi = g1[j]             + g2[j]             + bih[j]             + bhh[j];
  float gf = g1[j + DEC_H]     + g2[j + DEC_H]     + bih[j + DEC_H]     + bhh[j + DEC_H];
  float gg = g1[j + 2 * DEC_H] + g2[j + 2 * DEC_H] + bih[j + 2 * DEC_H] + bhh[j + 2 * DEC_H];
  float go = g1[j + 3 * DEC_H] + g2[j + 3 * DEC_H] + bih[j + 3 * DEC_H] + bhh[j + 3 * DEC_H];
  float cn = sigf(gf) * c[i] + sigf(gi) * tanhf(gg);
  float hn = sigf(go) * tanhf(cn);
  c[i] = cn;
  h[i] = (__bf16)hn;
  A_wo[(long)b * (2 * DEC_H) + DEC_H + j] = (__bf16)hn;
}

// Row-wise log_softmax over vocabulary (+ bias), one batch row per block
__global__ void k_logsoftmax(const float* __restrict__ logits,
                             const float* __restrict__ gb,
                             float* __restrict__ out, int step) {
  const int b = blockIdx.x;
  const int tid = threadIdx.x, nt = blockDim.x;   // 512
  __shared__ float red[512];
  __shared__ float s_stat;
  const float* row = logits + (long)b * V_P;
  float m = -1e30f;
  for (int v = tid; v < TRG_V; v += nt) m = fmaxf(m, row[v] + gb[v]);
  red[tid] = m; __syncthreads();
  for (int o = nt >> 1; o > 0; o >>= 1) {
    if (tid < o) red[tid] = fmaxf(red[tid], red[tid + o]);
    __syncthreads();
  }
  if (tid == 0) s_stat = red[0];
  __syncthreads();
  const float M = s_stat;
  __syncthreads();
  float s = 0.f;
  for (int v = tid; v < TRG_V; v += nt) s += __expf(row[v] + gb[v] - M);
  red[tid] = s; __syncthreads();
  for (int o = nt >> 1; o > 0; o >>= 1) {
    if (tid < o) red[tid] += red[tid + o];
    __syncthreads();
  }
  if (tid == 0) s_stat = M + __logf(red[0]);
  __syncthreads();
  const float L = s_stat;
  float* o = out + ((long)(step + 1) * BATCH + b) * TRG_V;
  for (int v = tid; v < TRG_V; v += nt) o[v] = row[v] + gb[v] - L;
}

// ---------------------------------------------------------------------------
// Host orchestration
// ---------------------------------------------------------------------------
static inline void launch_gemm(const __bf16* A, int lda, const __bf16* B, int ldb,
                               float* C, int ldc, int M, int N, int K, hipStream_t s) {
  dim3 grid((N / 16 + 3) / 4, M / 64);
  k_gemm_bf16<<<grid, 128, 0, s>>>(A, lda, B, ldb, C, ldc, N, K);
}
static inline int gblk(long n, int t) { return (int)((n + t - 1) / t); }

extern "C" void kernel_launch(void* const* d_in, const int* in_sizes, int n_in,
                              void* d_out, int out_size, void* d_ws, size_t ws_size,
                              hipStream_t stream) {
  (void)in_sizes; (void)n_in; (void)out_size; (void)ws_size;
  const int*   src      = (const int*)d_in[0];
  const int*   trg      = (const int*)d_in[1];
  const float* enc_emb  = (const float*)d_in[2];
  const float* dec_emb  = (const float*)d_in[3];
  const float* wihF     = (const float*)d_in[4];
  const float* whhF     = (const float*)d_in[5];
  const float* bihF     = (const float*)d_in[6];
  const float* bhhF     = (const float*)d_in[7];
  const float* wihB     = (const float*)d_in[8];
  const float* whhB     = (const float*)d_in[9];
  const float* bihB     = (const float*)d_in[10];
  const float* bhhB     = (const float*)d_in[11];
  const float* dWih     = (const float*)d_in[12];
  const float* dWhh     = (const float*)d_in[13];
  const float* dbih     = (const float*)d_in[14];
  const float* dbhh     = (const float*)d_in[15];
  const float* wi       = (const float*)d_in[16];
  const float* wo       = (const float*)d_in[17];
  const float* genW     = (const float*)d_in[18];
  const float* genB     = (const float*)d_in[19];
  float* out = (float*)d_out;

  // ---- workspace carve-up (256B aligned) ----
  char* base = (char*)d_ws;
  size_t off = 0;
  auto carve = [&](size_t bytes) -> char* {
    char* p = base + off;
    off = (off + bytes + 255) & ~(size_t)255;
    return p;
  };
  __bf16* Xsrc    = (__bf16*)carve((size_t)SRC_LEN * BATCH * EMB_P * 2);
  __bf16* wihF_b  = (__bf16*)carve((size_t)4 * ENC_H * EMB_P * 2);
  __bf16* wihB_b  = (__bf16*)carve((size_t)4 * ENC_H * EMB_P * 2);
  __bf16* whhF_b  = (__bf16*)carve((size_t)4 * ENC_H * ENC_H * 2);
  __bf16* whhB_b  = (__bf16*)carve((size_t)4 * ENC_H * ENC_H * 2);
  __bf16* dWih_b  = (__bf16*)carve((size_t)4 * DEC_H * K_DEC_IN * 2);
  __bf16* dWhh_b  = (__bf16*)carve((size_t)4 * DEC_H * DEC_H * 2);
  __bf16* wi_b    = (__bf16*)carve((size_t)DEC_H * DEC_H * 2);
  __bf16* wo_b    = (__bf16*)carve((size_t)DEC_H * 2 * DEC_H * 2);
  __bf16* genW_b  = (__bf16*)carve((size_t)V_P * DEC_H * 2);
  float*  XWf     = (float*)carve((size_t)SRC_LEN * BATCH * 4 * ENC_H * 4);
  float*  XWb     = (float*)carve((size_t)SRC_LEN * BATCH * 4 * ENC_H * 4);
  float*  enc_out = (float*)carve((size_t)SRC_LEN * BATCH * 2 * ENC_H * 4);
  float*  HWf     = (float*)carve((size_t)BATCH * 4 * ENC_H * 4);
  float*  HWb     = (float*)carve((size_t)BATCH * 4 * ENC_H * 4);
  __bf16* hF      = (__bf16*)carve((size_t)BATCH * ENC_H * 2);
  __bf16* hB      = (__bf16*)carve((size_t)BATCH * ENC_H * 2);
  float*  cF      = (float*)carve((size_t)BATCH * ENC_H * 4);
  float*  cB      = (float*)carve((size_t)BATCH * ENC_H * 4);
  float*  qbuf    = (float*)carve((size_t)BATCH * DEC_H * 4);
  __bf16* A_wo    = (__bf16*)carve((size_t)BATCH * 2 * DEC_H * 2);
  float*  ct_lin  = (float*)carve((size_t)BATCH * DEC_H * 4);
  __bf16* inp     = (__bf16*)carve((size_t)BATCH * K_DEC_IN * 2);
  float*  G1      = (float*)carve((size_t)BATCH * 4 * DEC_H * 4);
  float*  G2      = (float*)carve((size_t)BATCH * 4 * DEC_H * 4);
  __bf16* h_dec   = (__bf16*)carve((size_t)BATCH * DEC_H * 2);
  float*  c_dec   = (float*)carve((size_t)BATCH * DEC_H * 4);
  float*  logits  = (float*)carve((size_t)BATCH * V_P * 4);

  const int T = 256;
  // ---- phase 0: bf16 weight casts (with padding) ----
  k_cast_pad<<<gblk((long)4*ENC_H*EMB_P, T), T, 0, stream>>>(wihF, wihF_b, 4*ENC_H, EMB, 4*ENC_H, EMB_P);
  k_cast_pad<<<gblk((long)4*ENC_H*EMB_P, T), T, 0, stream>>>(wihB, wihB_b, 4*ENC_H, EMB, 4*ENC_H, EMB_P);
  k_cast_pad<<<gblk((long)4*ENC_H*ENC_H, T), T, 0, stream>>>(whhF, whhF_b, 4*ENC_H, ENC_H, 4*ENC_H, ENC_H);
  k_cast_pad<<<gblk((long)4*ENC_H*ENC_H, T), T, 0, stream>>>(whhB, whhB_b, 4*ENC_H, ENC_H, 4*ENC_H, ENC_H);
  k_cast_pad<<<gblk((long)4*DEC_H*K_DEC_IN, T), T, 0, stream>>>(dWih, dWih_b, 4*DEC_H, DEC_H+EMB, 4*DEC_H, K_DEC_IN);
  k_cast_pad<<<gblk((long)4*DEC_H*DEC_H, T), T, 0, stream>>>(dWhh, dWhh_b, 4*DEC_H, DEC_H, 4*DEC_H, DEC_H);
  k_cast_pad<<<gblk((long)DEC_H*DEC_H, T), T, 0, stream>>>(wi, wi_b, DEC_H, DEC_H, DEC_H, DEC_H);
  k_cast_pad<<<gblk((long)DEC_H*2*DEC_H, T), T, 0, stream>>>(wo, wo_b, DEC_H, 2*DEC_H, DEC_H, 2*DEC_H);
  k_cast_pad<<<gblk((long)V_P*DEC_H, T), T, 0, stream>>>(genW, genW_b, TRG_V, DEC_H, V_P, DEC_H);
  k_embed_src<<<gblk((long)SRC_LEN*BATCH*EMB_P, T), T, 0, stream>>>(src, enc_emb, Xsrc);

  // ---- state init ----
  k_fill_f32 <<<gblk((long)BATCH*ENC_H, T), T, 0, stream>>>(cF, (long)BATCH*ENC_H, 0.f);
  k_fill_f32 <<<gblk((long)BATCH*ENC_H, T), T, 0, stream>>>(cB, (long)BATCH*ENC_H, 0.f);
  k_fill_bf16<<<gblk((long)BATCH*ENC_H, T), T, 0, stream>>>(hF, (long)BATCH*ENC_H, 0.f);
  k_fill_bf16<<<gblk((long)BATCH*ENC_H, T), T, 0, stream>>>(hB, (long)BATCH*ENC_H, 0.f);
  const float init = 1.0f / 1024.0f;
  k_fill_f32 <<<gblk((long)BATCH*DEC_H, T), T, 0, stream>>>(c_dec, (long)BATCH*DEC_H, init);
  k_fill_bf16<<<gblk((long)BATCH*DEC_H, T), T, 0, stream>>>(h_dec, (long)BATCH*DEC_H, init);
  k_fill_bf16<<<gblk((long)BATCH*2*DEC_H, T), T, 0, stream>>>(A_wo, (long)BATCH*2*DEC_H, init);
  k_fill_f32 <<<gblk((long)BATCH*TRG_V, T), T, 0, stream>>>(out, (long)BATCH*TRG_V, 0.f); // result[0]=0

  // ---- encoder: hoisted input GEMMs, then 50 recurrent steps ----
  launch_gemm(Xsrc, EMB_P, wihF_b, EMB_P, XWf, 4*ENC_H, SRC_LEN*BATCH, 4*ENC_H, EMB_P, stream);
  launch_gemm(Xsrc, EMB_P, wihB_b, EMB_P, XWb, 4*ENC_H, SRC_LEN*BATCH, 4*ENC_H, EMB_P, stream);
  for (int t = 0; t < SRC_LEN; ++t) {
    launch_gemm(hF, ENC_H, whhF_b, ENC_H, HWf, 4*ENC_H, BATCH, 4*ENC_H, ENC_H, stream);
    k_enc_gate<<<gblk((long)BATCH*ENC_H, T), T, 0, stream>>>(
        XWf, t * BATCH, HWf, bihF, bhhF, cF, hF, enc_out, t, 0);
    const int tb = SRC_LEN - 1 - t;
    launch_gemm(hB, ENC_H, whhB_b, ENC_H, HWb, 4*ENC_H, BATCH, 4*ENC_H, ENC_H, stream);
    k_enc_gate<<<gblk((long)BATCH*ENC_H, T), T, 0, stream>>>(
        XWb, tb * BATCH, HWb, bihB, bhhB, cB, hB, enc_out, tb, ENC_H);
  }

  // ---- decoder: 39 steps ----
  for (int step = 0; step < TRG_LEN - 1; ++step) {
    launch_gemm(h_dec, DEC_H, wi_b, DEC_H, qbuf, DEC_H, BATCH, DEC_H, DEC_H, stream);
    k_attention<<<BATCH, 256, 0, stream>>>(enc_out, qbuf, A_wo);
    launch_gemm(A_wo, 2*DEC_H, wo_b, 2*DEC_H, ct_lin, DEC_H, BATCH, DEC_H, 2*DEC_H, stream);
    k_ct_inp<<<gblk((long)BATCH*K_DEC_IN, T), T, 0, stream>>>(ct_lin, dec_emb, trg, step, inp);
    launch_gemm(inp, K_DEC_IN, dWih_b, K_DEC_IN, G1, 4*DEC_H, BATCH, 4*DEC_H, K_DEC_IN, stream);
    launch_gemm(h_dec, DEC_H, dWhh_b, DEC_H, G2, 4*DEC_H, BATCH, 4*DEC_H, DEC_H, stream);
    k_dec_lstm<<<gblk((long)BATCH*DEC_H, T), T, 0, stream>>>(G1, G2, dbih, dbhh, c_dec, h_dec, A_wo);
    launch_gemm(h_dec, DEC_H, genW_b, DEC_H, logits, V_P, BATCH, V_P, DEC_H, stream);
    k_logsoftmax<<<BATCH, 512, 0, stream>>>(logits, genB, out, step);
  }
}
